// PerTokenFN_49185965474011
// MI455X (gfx1250) — compile-verified, matrix operand order checked by reading, not run on hardware
//
#include <hip/hip_runtime.h>
#include <hip/hip_bf16.h>
#include <math.h>

// Grouped per-position MLP:  out[b,t,:] = W2[t]^T gelu(W1[t]^T x[b,t,:] + b1[t]) + b2[t]
// B=4096, T=64, D=256, H=1024.  bf16 WMMA multiplies, fp32 accumulate, exact-erf GELU.
// LDS tiles are stored pre-swizzled in WMMA-fragment layout so every fragment load
// is exactly two ds_load_b128 per lane.

typedef __attribute__((ext_vector_type(16))) __bf16    v16bf;
typedef __attribute__((ext_vector_type(8)))  float     v8f;
typedef __attribute__((ext_vector_type(4)))  unsigned  v4u;

#define Bsz 4096
#define Tsz 64
#define Dsz 256
#define Hsz 1024
#define BM  64            // token rows per workgroup
#define CH  64            // hidden chunk per iteration
#define NCHUNK (Hsz / CH) // 16

union Frag { v4u q[2]; unsigned u[8]; v16bf v; };

__device__ __forceinline__ unsigned short f2bf(float f) {
  union { float f; unsigned u; } x; x.f = f;
  unsigned r = x.u + 0x7FFFu + ((x.u >> 16) & 1u);   // round-to-nearest-even
  return (unsigned short)(r >> 16);
}

// Fragment-native swizzle (ISA 16-bit A 16x32 layout; B uses A-layout of B^T):
//   lane = low16 + 16*((kin>>3)&1),  elem = (kin&7) + 8*(kin>>4)
// A fragment occupies 32 lanes x 16 bf16 contiguous per lane (1 KB per tile).
__device__ __forceinline__ v16bf loadFrag(const unsigned short* base, int tile, int lane) {
  Frag f;
  const v4u* p = (const v4u*)(base + (((size_t)tile * 32 + lane) << 4));
  f.q[0] = p[0];          // elements 0..7   -> ds_load_b128
  f.q[1] = p[1];          // elements 8..15  -> ds_load_b128
  return f.v;
}

// Stage a KR x NC fp32 chunk (global row stride GS) into swizzled bf16 tiles.
// Tile grid: tile = kt*(NC/16) + ntile. Reads row-pairs so each store is ds_store_b32.
template <int KR, int NC, int GS>
__device__ __forceinline__ void stageW(const float* __restrict__ src,
                                       unsigned short* dst, int tid) {
  constexpr int NQ   = NC / 4;               // float4 per row
  constexpr int ITER = (KR / 2) * NQ / 256;  // 8 for both W1 and W2 chunks
#pragma unroll
  for (int i = 0; i < ITER; ++i) {
    int widx = tid + i * 256;
    int kp = widx / NQ;                      // row pair index
    int nq = widx % NQ;
    int k  = kp * 2;
    int n0 = nq * 4;
    float4 a = *(const float4*)(src + (size_t)k       * GS + n0);
    float4 b = *(const float4*)(src + (size_t)(k + 1) * GS + n0);
    int kt    = k >> 5;
    int kin   = k & 31;
    int laneK = 16 * ((kin >> 3) & 1);
    int e     = (kin & 7) + 8 * (kin >> 4);  // even; k+1 -> e+1 (same lane)
    float av[4] = {a.x, a.y, a.z, a.w};
    float bv[4] = {b.x, b.y, b.z, b.w};
#pragma unroll
    for (int j = 0; j < 4; ++j) {
      int n    = n0 + j;
      int tile = kt * (NC / 16) + (n >> 4);
      unsigned short* p = dst + (((size_t)tile * 32 + laneK + (n & 15)) << 4) + e;
      *(unsigned*)p = (unsigned)f2bf(av[j]) | ((unsigned)f2bf(bv[j]) << 16);
    }
  }
}

__global__ __launch_bounds__(256, 2)
void PerTokenFN_mlp_wmma_kernel(const float* __restrict__ x,
                                const float* __restrict__ W1,
                                const float* __restrict__ b1,
                                const float* __restrict__ W2,
                                const float* __restrict__ b2,
                                float* __restrict__ out)
{
  // 104 KB LDS — exploits CDNA5's 320 KB WGP-local LDS
  __shared__ __align__(16) unsigned short xs [BM * Dsz];  // 64x256 bf16 : 32 KB (swizzled)
  __shared__ __align__(16) unsigned short w1s[Dsz * CH];  // 256x64 bf16 : 32 KB (swizzled)
  __shared__ __align__(16) unsigned short w2s[CH * Dsz];  // 64x256 bf16 : 32 KB (swizzled)
  __shared__ __align__(16) unsigned short hs [BM * CH];   // 64x64  bf16 :  8 KB (swizzled)

  const int tid  = threadIdx.x;
  const int lane = tid & 31;
  const int w    = tid >> 5;   // 8 wave32s
  const int wm   = w >> 1;     // row strip 0..3  (16 rows each)
  const int wn   = w & 1;      // col half  0..1
  const int b0   = blockIdx.x * BM;
  const int t    = blockIdx.y;

  // ---- stage x tile (fp32 -> bf16, swizzled): 64 rows(m) x 256 cols(k) ----
#pragma unroll
  for (int i = 0; i < 16; ++i) {
    int idx = tid + i * 256;        // float4 index, 4096 total
    int m   = idx >> 6;             // 64 float4 per row
    int k0  = (idx & 63) << 2;      // 4 consecutive K, same lane & elem group
    float4 v = *(const float4*)(x + ((size_t)(b0 + m) * Tsz + t) * Dsz + k0);
    int kt    = k0 >> 5;
    int kin   = k0 & 31;
    int lane2 = (m & 15) + 16 * ((kin >> 3) & 1);
    int e     = (kin & 7) + 8 * (kin >> 4);            // multiple of 4
    int tile  = kt * (BM / 16) + (m >> 4);
    unsigned short* p = xs + (((size_t)tile * 32 + lane2) << 4) + e;
    unsigned lo = (unsigned)f2bf(v.x) | ((unsigned)f2bf(v.y) << 16);
    unsigned hi = (unsigned)f2bf(v.z) | ((unsigned)f2bf(v.w) << 16);
    *(uint2*)p = make_uint2(lo, hi);                   // one ds_store_b64
  }

  v8f oacc[8] = {};                 // 16 x 128 fp32 output accumulators per wave

  for (int hc = 0; hc < NCHUNK; ++hc) {
    __syncthreads();                // protect w1s/w2s/hs reuse across iterations

    // ---- stage W1[:, chunk] (256x64, row stride H) and W2[chunk, :] (64x256) ----
    stageW<Dsz, CH, Hsz>(W1 + (size_t)t * Dsz * Hsz + (size_t)hc * CH,       w1s, tid);
    stageW<CH, Dsz, Dsz>(W2 + (size_t)t * Hsz * Dsz + (size_t)hc * CH * Dsz, w2s, tid);

    // L2 prefetch of next W2 chunk (contiguous 64 KB) -> global_prefetch_b8
    if (hc + 1 < NCHUNK) {
      const float* nw2 = W2 + (size_t)t * Hsz * Dsz + (size_t)(hc + 1) * CH * Dsz;
      __builtin_prefetch(nw2 + (size_t)tid * 64, 0, 1);
    }
    __syncthreads();

    // ---- GEMM1: h(16 x 32) = x(16 x 256) @ W1chunk(256 x 32-of-64) ----
    v8f hacc[2] = {};
#pragma unroll
    for (int kk = 0; kk < Dsz / 32; ++kk) {
      v16bf a = loadFrag(xs, kk * (BM / 16) + wm, lane);
#pragma unroll
      for (int nt = 0; nt < 2; ++nt) {
        v16bf bb = loadFrag(w1s, kk * (CH / 16) + wn * 2 + nt, lane);
        hacc[nt] = __builtin_amdgcn_wmma_f32_16x16x32_bf16(
            false, a, false, bb, (short)0, hacc[nt], false, false);
      }
    }

    // ---- bias + exact GELU, write bf16 into swizzled hs (C-layout -> A-layout) ----
#pragma unroll
    for (int nt = 0; nt < 2; ++nt) {
      int kcol = wn * 32 + nt * 16 + (lane & 15);      // K index within 64-chunk
      float bias = b1[t * Hsz + hc * CH + kcol];
      int kt    = kcol >> 5;
      int kin   = kcol & 31;
      int laneK = 16 * ((kin >> 3) & 1);
      int e     = (kin & 7) + 8 * (kin >> 4);
#pragma unroll
      for (int r = 0; r < 8; ++r) {
        int m = wm * 16 + r + 8 * (lane >> 4);
        float v = hacc[nt][r] + bias;
        v = 0.5f * v * (1.0f + erff(v * 0.70710678118654752f));  // exact erf GELU
        int tile = kt * (BM / 16) + (m >> 4);
        hs[(((size_t)tile * 32 + laneK + (m & 15)) << 4) + e] = f2bf(v);
      }
    }
    __syncthreads();

    // ---- GEMM2: out(16 x 128) += h(16 x 64) @ W2chunk(64 x 128-of-256) ----
#pragma unroll
    for (int kk = 0; kk < CH / 32; ++kk) {
      v16bf a = loadFrag(hs, kk * (BM / 16) + wm, lane);
#pragma unroll
      for (int nt = 0; nt < 8; ++nt) {
        v16bf bb = loadFrag(w2s, kk * (Dsz / 16) + wn * 8 + nt, lane);
        oacc[nt] = __builtin_amdgcn_wmma_f32_16x16x32_bf16(
            false, a, false, bb, (short)0, oacc[nt], false, false);
      }
    }
  }

  // ---- epilogue: + b2, store fp32 ----
#pragma unroll
  for (int nt = 0; nt < 8; ++nt) {
    int col = wn * 128 + nt * 16 + (lane & 15);
    float bias = b2[t * Dsz + col];
#pragma unroll
    for (int r = 0; r < 8; ++r) {
      int m   = r + 8 * (lane >> 4);
      int row = b0 + wm * 16 + m;
      out[((size_t)row * Tsz + t) * Dsz + col] = oacc[nt][r] + bias;
    }
  }
}

extern "C" void kernel_launch(void* const* d_in, const int* in_sizes, int n_in,
                              void* d_out, int out_size, void* d_ws, size_t ws_size,
                              hipStream_t stream) {
  (void)in_sizes; (void)n_in; (void)out_size; (void)d_ws; (void)ws_size;
  const float* x  = (const float*)d_in[0];
  const float* W1 = (const float*)d_in[1];
  const float* b1 = (const float*)d_in[2];
  const float* W2 = (const float*)d_in[3];
  const float* b2 = (const float*)d_in[4];
  float* out = (float*)d_out;

  dim3 grid(Bsz / BM, Tsz);   // 64 token-tiles x 64 positions
  dim3 block(256);            // 8 wave32s
  PerTokenFN_mlp_wmma_kernel<<<grid, block, 0, stream>>>(x, W1, b1, W2, b2, out);
}
